// VAE_5927054868983
// MI455X (gfx1250) — matrix operand order repaired and checked by reading, not compiled, not run on hardware
//
#include <hip/hip_runtime.h>
#include <hip/hip_bf16.h>
#include <stdint.h>

#define H      2048
#define G4     8192        // 4*H (gates)
#define T_IN   512
#define T_OUT  512
#define NB_LSTM 64
#define BT_LSTM 128

typedef __attribute__((ext_vector_type(16))) __bf16   v16bf;
typedef __attribute__((ext_vector_type(8)))  float    v8f;
typedef __attribute__((ext_vector_type(4)))  unsigned u32x4;
typedef __attribute__((ext_vector_type(8)))  unsigned u32x8;

// ---------- helpers ----------
__device__ inline unsigned short f2bf(float f) {
  unsigned u = __builtin_bit_cast(unsigned, f);
  u += 0x7fffu + ((u >> 16) & 1u);          // round-to-nearest-even
  return (unsigned short)(u >> 16);
}
__device__ inline float bflo(unsigned p) { return __builtin_bit_cast(float, p << 16); }
__device__ inline float bfhi(unsigned p) { return __builtin_bit_cast(float, p & 0xffff0000u); }
__device__ inline float sigmoidf(float x) { return 1.0f / (1.0f + __expf(-x)); }

// raw 32-bit LDS offset of a generic pointer that aliases LDS
__device__ inline unsigned lds_addr32(const void* p) {
  return (unsigned)(unsigned long long)(const __attribute__((address_space(3))) char*)p;
}
// CDNA5 async memory->LDS DMA (ASYNCcnt-tracked), 16B per lane
__device__ inline void async_ld128(unsigned ldsa, const void* g) {
  asm volatile("global_load_async_to_lds_b128 %0, %1, off"
               :: "v"(ldsa), "v"((unsigned long long)(uintptr_t)g) : "memory");
}
__device__ inline void wait_async0() { asm volatile("s_wait_asynccnt 0x0" ::: "memory"); }
__device__ inline void wait_async4() { asm volatile("s_wait_asynccnt 0x4" ::: "memory"); }

// ---------- f32 -> packed bf16 pairs ----------
__global__ void k_cvt_bf16(const float* __restrict__ src, unsigned* __restrict__ dst, int npairs) {
  int i = blockIdx.x * blockDim.x + threadIdx.x;
  int stride = gridDim.x * blockDim.x;
  for (; i < npairs; i += stride) {
    float2 f = ((const float2*)src)[i];
    dst[i] = (unsigned)f2bf(f.x) | ((unsigned)f2bf(f.y) << 16);
  }
}

// ---------- embedding gather -> bf16 activation matrix [T][H] ----------
__global__ void k_embed(const int* __restrict__ toks, const float* __restrict__ emb,
                        unsigned short* __restrict__ Abf, int T, int shifted) {
  int i = blockIdx.x * blockDim.x + threadIdx.x;
  int stride = gridDim.x * blockDim.x;
  int total = T * H;
  for (; i < total; i += stride) {
    int t = i >> 11;                 // /H
    int k = i & (H - 1);
    int tok = shifted ? (t == 0 ? 0 : toks[t - 1]) : toks[t];
    Abf[i] = f2bf(emb[(size_t)tok * H + k]);
  }
}

// ---------- WMMA bf16 GEMM: xproj[T][8192] = A[T][2048] * W^T + bih ----------
// Block = one tm (16 rows) x 8 tn tiles (one per wave). A staged into LDS with
// double-buffered async-to-LDS DMA (4 sections of K=512); B streamed from L2.
#define KSEC 512
#define NSEC (H / KSEC)            // 4
__global__ void k_xproj_wmma(const unsigned short* __restrict__ Abf,
                             const unsigned short* __restrict__ Wbf,
                             const float* __restrict__ bih,
                             float* __restrict__ xproj, int tilesM) {
  __shared__ __align__(16) unsigned short Ast[2][16 * KSEC];   // 2 x 16KB
  const int tid  = threadIdx.x;
  const int lane = tid & 31;
  const int wid  = tid >> 5;                  // 0..7
  const int grp  = blockIdx.x & 63;           // tn group (512/8 tiles)
  const int tm   = blockIdx.x >> 6;
  if (tm >= tilesM) return;                   // uniform; grid sized exactly
  const int tn   = grp * 8 + wid;
  const int half = lane >> 4;
  const int mn   = lane & 15;

  const u32x4* bp = (const u32x4*)(Wbf + (size_t)(tn * 16 + mn) * H);
  const u32x4* ap = (const u32x4*)&Ast[0][mn * KSEC];          // buffer 0 base
  const int    bufStride = (16 * KSEC) / 8;                    // u32x4 per buffer

  // stage section s into buffer b: 16 rows x 1KB, 1024 16B chunks, 4 per thread
  auto stage = [&](int s, int b) {
    #pragma unroll
    for (int i = 0; i < 4; ++i) {
      int cch = tid + 256 * i;
      int row = cch >> 6, col = cch & 63;     // 64 chunks per row
      const char* gp = (const char*)Abf +
          (((size_t)(tm * 16 + row)) * H + (size_t)s * KSEC + (size_t)col * 8) * 2;
      unsigned lo = lds_addr32(&Ast[b][row * KSEC + col * 8]);
      async_ld128(lo, gp);
    }
  };

  v8f acc = {0.f, 0.f, 0.f, 0.f, 0.f, 0.f, 0.f, 0.f};

  stage(0, 0);
  for (int s = 0; s < NSEC; ++s) {
    if (s + 1 < NSEC) { stage(s + 1, (s + 1) & 1); wait_async4(); }
    else              { wait_async0(); }
    __syncthreads();                           // all waves' section s staged

    const u32x4* abuf = ap + (s & 1) * bufStride;
    #pragma unroll
    for (int kcl = 0; kcl < KSEC / 32; ++kcl) {
      // A 16x32 bf16 from LDS: lane holds K = k0+8*half+[0,8) and k0+16+8*half+[0,8)
      u32x4 a0 = abuf[kcl * 4 + half];
      u32x4 a1 = abuf[kcl * 4 + half + 2];
      // B 32x16 bf16 from global: lane (col) holds K = K0+8*half+[0,8) and K0+16+8*half+[0,8)
      const int q = (s * (KSEC / 32) + kcl) * 4;
      u32x4 b0 = bp[q + half];
      u32x4 b1 = bp[q + half + 2];
      u32x8 ar = {a0.x, a0.y, a0.z, a0.w, a1.x, a1.y, a1.z, a1.w};
      u32x8 br = {b0.x, b0.y, b0.z, b0.w, b1.x, b1.y, b1.z, b1.w};
      v16bf av = __builtin_bit_cast(v16bf, ar);
      v16bf bv = __builtin_bit_cast(v16bf, br);
      acc = __builtin_amdgcn_wmma_f32_16x16x32_bf16(false, av, false, bv,
                                                    (short)0, acc, false, false);
    }
    __syncthreads();                           // done reading before restage
  }

  const int nq = tn * 16 + mn;
  const float bias = bih[nq];
  #pragma unroll
  for (int v = 0; v < 8; ++v) {
    int m = v + 8 * half;                      // C layout: lane=N, vgpr v -> M
    xproj[(size_t)(tm * 16 + m) * G4 + nq] = acc[v] + bias;
  }
}

// ---------- init state / barrier ----------
__global__ void k_init(float* h, float* c, unsigned short* hbf, unsigned* bar, int full) {
  int i = threadIdx.x;
  if (i < 8) bar[i] = 0u;
  if (full) {
    for (int j = i; j < H; j += blockDim.x) { h[j] = 0.f; c[j] = 0.f; hbf[j] = 0; }
  }
}

// ---------- device-wide sense barrier ----------
__device__ inline void gbar(unsigned* cnt, unsigned* gen, unsigned nb) {
  __syncthreads();
  __threadfence();
  if (threadIdx.x == 0) {
    unsigned g = __hip_atomic_load(gen, __ATOMIC_RELAXED, __HIP_MEMORY_SCOPE_AGENT);
    unsigned a = __hip_atomic_fetch_add(cnt, 1u, __ATOMIC_ACQ_REL, __HIP_MEMORY_SCOPE_AGENT);
    if (a == nb - 1u) {
      __hip_atomic_store(cnt, 0u, __ATOMIC_RELAXED, __HIP_MEMORY_SCOPE_AGENT);
      __hip_atomic_fetch_add(gen, 1u, __ATOMIC_RELEASE, __HIP_MEMORY_SCOPE_AGENT);
    } else {
      while (__hip_atomic_load(gen, __ATOMIC_ACQUIRE, __HIP_MEMORY_SCOPE_AGENT) == g)
        __builtin_amdgcn_s_sleep(2);
    }
  }
  __threadfence();
  __syncthreads();
}

// ---------- persistent LSTM: 512 sequential steps, Whh bf16 out of L2 ----------
// 64 blocks x 128 threads; thread = one gate row (q*H + j); c kept in registers.
__global__ void k_lstm(const float* __restrict__ xproj,
                       const unsigned short* __restrict__ Whh,
                       const float* __restrict__ bhh,
                       float* __restrict__ gates,
                       float* __restrict__ h, unsigned short* __restrict__ hbf,
                       float* __restrict__ c, float* __restrict__ outs,
                       unsigned* bar, int T) {
  __shared__ __align__(16) unsigned lds_h[H / 2];   // h as packed bf16 pairs (4KB)
  const int tid = threadIdx.x;
  const int q   = tid >> 5;                       // gate quadrant (i,f,g,o)
  const int jj  = tid & 31;
  const int j   = blockIdx.x * 32 + jj;           // hidden index owned
  const int row = q * H + j;                      // Whh row / gate index
  const float brow = bhh[row];
  const u32x4* wp = (const u32x4*)(Whh + (size_t)row * H);
  float creg = (tid < 32) ? c[j] : 0.f;

  const unsigned hlds = lds_addr32(&lds_h[tid * 8]);   // 32B per thread
  for (int t = 0; t < T; ++t) {
    // async DMA h (bf16, 4KB) into LDS: 2 x b128 per thread
    const char* gp = (const char*)hbf + (size_t)tid * 32;
    async_ld128(hlds, gp);
    async_ld128(hlds + 16, gp + 16);
    wait_async0();
    __syncthreads();

    float acc = 0.f;
    #pragma unroll 4
    for (int kk = 0; kk < H / 8; ++kk) {          // 8 bf16 per iter
      if ((kk & 31) == 0) __builtin_prefetch(&wp[kk + 32], 0, 0);
      u32x4 w = wp[kk];
      unsigned ha = lds_h[kk * 4 + 0];
      unsigned hb = lds_h[kk * 4 + 1];
      unsigned hc = lds_h[kk * 4 + 2];
      unsigned hd = lds_h[kk * 4 + 3];
      acc = fmaf(bflo(w.x), bflo(ha), acc);
      acc = fmaf(bfhi(w.x), bfhi(ha), acc);
      acc = fmaf(bflo(w.y), bflo(hb), acc);
      acc = fmaf(bfhi(w.y), bfhi(hb), acc);
      acc = fmaf(bflo(w.z), bflo(hc), acc);
      acc = fmaf(bfhi(w.z), bfhi(hc), acc);
      acc = fmaf(bflo(w.w), bflo(hd), acc);
      acc = fmaf(bfhi(w.w), bfhi(hd), acc);
    }
    gates[row] = acc + xproj[(size_t)t * G4 + row] + brow;

    gbar(bar + 0, bar + 1, gridDim.x);

    if (tid < 32) {
      float gi = sigmoidf(gates[j]);
      float gf = sigmoidf(gates[H + j]);
      float gg = tanhf(gates[2 * H + j]);
      float go = sigmoidf(gates[3 * H + j]);
      creg = gf * creg + gi * gg;
      float hv = go * tanhf(creg);
      h[j]  = hv;
      hbf[j] = f2bf(hv);
      if (outs) outs[(size_t)t * H + j] = hv;
    }
    gbar(bar + 0, bar + 1, gridDim.x);
  }
  if (tid < 32) c[j] = creg;                      // final c (feeds decoder)
}

// ---------- mu / logvar heads (f32 GEMV) ----------
__global__ void k_muz(const float* __restrict__ henc,
                      const float* __restrict__ muW, const float* __restrict__ mub,
                      const float* __restrict__ lvW, const float* __restrict__ lvb,
                      float* __restrict__ mu_out, float* __restrict__ lv_out) {
  int j = blockIdx.x * blockDim.x + threadIdx.x;
  if (j >= H) return;
  const float4* hp = (const float4*)henc;
  const float4* mp = (const float4*)(muW + (size_t)j * H);
  const float4* lp = (const float4*)(lvW + (size_t)j * H);
  float am = 0.f, al = 0.f;
  for (int k = 0; k < H / 4; ++k) {
    float4 hv = hp[k], m = mp[k], l = lp[k];
    am += hv.x * m.x + hv.y * m.y + hv.z * m.z + hv.w * m.w;
    al += hv.x * l.x + hv.y * l.y + hv.z * l.z + hv.w * l.w;
  }
  mu_out[j] = am + mub[j];
  lv_out[j] = al + lvb[j];
}

// ---------- reparameterize: z -> h buffers for decoder ----------
__global__ void k_z(const float* __restrict__ mu, const float* __restrict__ lv,
                    const float* __restrict__ eps,
                    float* __restrict__ h, unsigned short* __restrict__ hbf) {
  int j = blockIdx.x * blockDim.x + threadIdx.x;
  if (j >= H) return;
  float z = mu[j] + eps[j] * __expf(0.5f * lv[j]);
  h[j] = z;
  hbf[j] = f2bf(z);
}

// ---------- argmax over hidden dim (first-max wins) ----------
__global__ void k_argmax(const float* __restrict__ outs, float* __restrict__ res) {
  __shared__ float sv[256];
  __shared__ int   si[256];
  const int t = blockIdx.x;
  const float* row = outs + (size_t)t * H;
  float best = -3.402823466e38f; int bi = 0;
  for (int k = threadIdx.x; k < H; k += 256) {
    float v = row[k];
    if (v > best) { best = v; bi = k; }
  }
  sv[threadIdx.x] = best; si[threadIdx.x] = bi;
  __syncthreads();
  for (int s = 128; s > 0; s >>= 1) {
    if (threadIdx.x < (unsigned)s) {
      float ov = sv[threadIdx.x + s]; int oi = si[threadIdx.x + s];
      if (ov > sv[threadIdx.x] || (ov == sv[threadIdx.x] && oi < si[threadIdx.x])) {
        sv[threadIdx.x] = ov; si[threadIdx.x] = oi;
      }
    }
    __syncthreads();
  }
  if (threadIdx.x == 0) res[t] = (float)si[0];
}

// =====================================================================
extern "C" void kernel_launch(void* const* d_in, const int* in_sizes, int n_in,
                              void* d_out, int out_size, void* d_ws, size_t ws_size,
                              hipStream_t stream) {
  (void)in_sizes; (void)n_in; (void)out_size; (void)ws_size;

  const int*   x       = (const int*)  d_in[0];
  const int*   target  = (const int*)  d_in[1];
  const float* eps     = (const float*)d_in[2];
  const float* enc_emb = (const float*)d_in[3];
  const float* enc_Wih = (const float*)d_in[4];
  const float* enc_Whh = (const float*)d_in[5];
  const float* enc_bih = (const float*)d_in[6];
  const float* enc_bhh = (const float*)d_in[7];
  const float* mu_W    = (const float*)d_in[8];
  const float* mu_b    = (const float*)d_in[9];
  const float* lv_W    = (const float*)d_in[10];
  const float* lv_b    = (const float*)d_in[11];
  const float* dec_emb = (const float*)d_in[12];
  const float* dec_Wih = (const float*)d_in[13];
  const float* dec_Whh = (const float*)d_in[14];
  const float* dec_bih = (const float*)d_in[15];
  const float* dec_bhh = (const float*)d_in[16];

  // d_out layout (floats): result[512] | dec_outs[512*2048] | mu[2048] | logvar[2048]
  float* out    = (float*)d_out;
  float* res    = out;
  float* decout = out + T_OUT;
  float* mu_out = out + T_OUT + (size_t)T_OUT * H;
  float* lv_out = mu_out + H;

  // workspace carve-up
  const size_t SZ_W = (size_t)G4 * H;                 // elements in one big weight
  unsigned short* wihE = (unsigned short*)d_ws;
  unsigned short* whhE = wihE + SZ_W;
  unsigned short* wihD = whhE + SZ_W;
  unsigned short* whhD = wihD + SZ_W;
  unsigned short* Aenc = whhD + SZ_W;                 // T_IN*H bf16
  unsigned short* Adec = Aenc + (size_t)T_IN * H;
  float* xprojE = (float*)(Adec + (size_t)T_OUT * H);
  float* xprojD = xprojE + (size_t)T_IN * G4;
  float* gates  = xprojD + (size_t)T_OUT * G4;
  float* hbuf   = gates + G4;
  float* cbuf   = hbuf + H;
  unsigned short* hbf = (unsigned short*)(cbuf + H);
  unsigned* bar = (unsigned*)(hbf + H);

  const int npairs = (int)(SZ_W / 2);

  // 1) weights -> bf16 (read once from HBM; thereafter L2-resident)
  k_cvt_bf16<<<4096, 256, 0, stream>>>(enc_Wih, (unsigned*)wihE, npairs);
  k_cvt_bf16<<<4096, 256, 0, stream>>>(enc_Whh, (unsigned*)whhE, npairs);
  k_cvt_bf16<<<4096, 256, 0, stream>>>(dec_Wih, (unsigned*)wihD, npairs);
  k_cvt_bf16<<<4096, 256, 0, stream>>>(dec_Whh, (unsigned*)whhD, npairs);

  // 2) token embeddings -> bf16 A matrices
  k_embed<<<1024, 256, 0, stream>>>(x,      enc_emb, Aenc, T_IN,  0);
  k_embed<<<1024, 256, 0, stream>>>(target, dec_emb, Adec, T_OUT, 1);   // [SOS]+target[:-1]

  // 3) input projections: WMMA bf16, A staged via double-buffered async-to-LDS
  k_xproj_wmma<<<(T_IN / 16) * 64, 256, 0, stream>>>(Aenc, wihE, enc_bih, xprojE, T_IN / 16);
  k_xproj_wmma<<<(T_OUT / 16) * 64, 256, 0, stream>>>(Adec, wihD, dec_bih, xprojD, T_OUT / 16);

  // 4) encoder recurrence (persistent, device-wide barriers)
  k_init<<<1, 256, 0, stream>>>(hbuf, cbuf, hbf, bar, 1);
  k_lstm<<<NB_LSTM, BT_LSTM, 0, stream>>>(xprojE, whhE, enc_bhh, gates,
                                          hbuf, hbf, cbuf, (float*)nullptr, bar, T_IN);

  // 5) VAE heads + reparameterize (z into h buffers; cbuf already holds c_enc)
  k_muz<<<16, 128, 0, stream>>>(hbuf, mu_W, mu_b, lv_W, lv_b, mu_out, lv_out);
  k_z  <<<16, 128, 0, stream>>>(mu_out, lv_out, eps, hbuf, hbf);

  // 6) decoder recurrence (writes dec_outs straight into d_out)
  k_init<<<1, 256, 0, stream>>>(hbuf, cbuf, hbf, bar, 0);   // barriers only
  k_lstm<<<NB_LSTM, BT_LSTM, 0, stream>>>(xprojD, whhD, dec_bhh, gates,
                                          hbuf, hbf, cbuf, decout, bar, T_OUT);

  // 7) argmax over hidden dim
  k_argmax<<<T_OUT, 256, 0, stream>>>(decout, res);
}
